// GCNNet_28235115004171
// MI455X (gfx1250) — compile-verified, matrix operand order checked by reading, not run on hardware
//
#include <hip/hip_runtime.h>
#include <math.h>

typedef __attribute__((ext_vector_type(16))) _Float16 v16h;
typedef __attribute__((ext_vector_type(8)))  _Float16 v8h;
typedef __attribute__((ext_vector_type(8)))  float    v8f;

#define D_HID 128
#define BASIS 64
#define KC    50
#define NCLS  10
#define NGRAPH 64

#define LDS_STRIDE 136   // halves; 272 B row stride -> 4-bank step per column

// ---------------------------------------------------------------------------
// WMMA GEMM:  H[n_rows x KOUT] = A[n_rows x 128] @ W[128 x KOUT] (+bias)(relu)
//  * Block = 256 threads (8 waves); each wave owns a 16-row tile and all
//    KOUT/16 column tiles (A fragment reused across column tiles).
//  * W is staged transposed into LDS as f16 once per block; B fragments are
//    two aligned ds_load_b128 per tile.
//  * f16 inputs, f32 accumulation: v_wmma_f32_16x16x32_f16, K = 4 x 32.
// ---------------------------------------------------------------------------
template <int KOUT>
__global__ void gemm_wmma_f16(const float* __restrict__ A,
                              const float* __restrict__ W,
                              const float* __restrict__ bias,
                              float* __restrict__ H,
                              int n_rows, int do_relu)
{
    constexpr int NT = KOUT / 16;            // column tiles per wave
    __shared__ _Float16 ldsW[KOUT * LDS_STRIDE];

    // ---- stage W^T into LDS as f16 (coalesced global reads) ----
    for (int idx = threadIdx.x; idx < 128 * KOUT; idx += 256) {
        int k = idx / KOUT;                  // W row   (K index)
        int n = idx % KOUT;                  // W col   (N index)
        ldsW[n * LDS_STRIDE + k] = (_Float16)W[idx];
    }
    __syncthreads();

    const int wave = threadIdx.x >> 5;
    const int lane = threadIdx.x & 31;
    const int m    = lane & 15;
    const int grp  = lane >> 4;              // 0: K-run base, 1: +8

    const int row0 = blockIdx.x * 128 + wave * 16;
    int row  = row0 + m;
    int rowc = row < n_rows ? row : (n_rows - 1);   // clamp: EXEC stays full
    const float* __restrict__ arow = A + (size_t)rowc * D_HID;

    v8f acc[NT];
    #pragma unroll
    for (int t = 0; t < NT; ++t) acc[t] = (v8f){};

    #pragma unroll
    for (int kk = 0; kk < 4; ++kk) {         // K = 128 = 4 x 32
        const int ka = kk * 32 + grp * 8;

        // A fragment (ISA §7.12.2 16-bit A): halves 0..7 -> K=ka+j,
        //                                    halves 8..15 -> K=ka+16+j
        v16h afrag;
        #pragma unroll
        for (int j = 0; j < 8; ++j) {
            afrag[j]     = (_Float16)arow[ka + j];
            afrag[j + 8] = (_Float16)arow[ka + 16 + j];
        }

        #pragma unroll
        for (int ct = 0; ct < NT; ++ct) {
            const int n = ct * 16 + m;       // B column for this lane
            const _Float16* bp = &ldsW[n * LDS_STRIDE + ka];
            v8h b0 = *(const v8h*)(bp);      // ds_load_b128 (16B aligned)
            v8h b1 = *(const v8h*)(bp + 16); // ds_load_b128
            v16h bfrag;
            #pragma unroll
            for (int j = 0; j < 8; ++j) { bfrag[j] = b0[j]; bfrag[j + 8] = b1[j]; }

            acc[ct] = __builtin_amdgcn_wmma_f32_16x16x32_f16(
                          false, afrag, false, bfrag,
                          (short)0, acc[ct], false, false);
        }
    }

    // D layout: VGPR r -> M = r + 8*grp, N = lane&15
    #pragma unroll
    for (int ct = 0; ct < NT; ++ct) {
        const int n_d = ct * 16 + m;
        const float bv = bias ? bias[n_d] : 0.0f;
        #pragma unroll
        for (int r = 0; r < 8; ++r) {
            int m_d = row0 + r + 8 * grp;
            if (m_d < n_rows) {
                float v = acc[ct][r] + bv;
                if (do_relu) v = fmaxf(v, 0.0f);
                H[(size_t)m_d * KOUT + n_d] = v;
            }
        }
    }
}

// ---------------------------------------------------------------------------
// Elementwise / graph kernels
// ---------------------------------------------------------------------------
__global__ void zero_f32(float* __restrict__ p, long long n)
{
    long long i = (long long)blockIdx.x * blockDim.x + threadIdx.x;
    if (i < n) p[i] = 0.0f;
}

__global__ void deg_count(const int* __restrict__ dst, float* __restrict__ deg, int E)
{
    int e = blockIdx.x * blockDim.x + threadIdx.x;
    if (e < E) atomicAdd(&deg[dst[e]], 1.0f);
}

__global__ void deg_rsqrt(const float* __restrict__ deg, float* __restrict__ dinv, int n)
{
    int i = blockIdx.x * blockDim.x + threadIdx.x;
    if (i < n) dinv[i] = rsqrtf(deg[i] + 1.0f);   // +1 self loop
}

// One wave per edge; each lane moves a 4-float chunk of the 128-wide message.
// h (51 MB) and agg (51 MB) both fit in the 192 MB L2, so gathers + f32
// atomics resolve on-chip, not in HBM.
__global__ void edge_msg(const int* __restrict__ src, const int* __restrict__ dst,
                         const float* __restrict__ dinv,
                         const float* __restrict__ h, float* __restrict__ agg, int E)
{
    long long tid = (long long)blockIdx.x * blockDim.x + threadIdx.x;
    int e = (int)(tid >> 5);
    if (e >= E) return;
    int c = (int)(tid & 31) * 4;
    int s = src[e], d = dst[e];
    float norm = dinv[s] * dinv[d];
    const float4 hv = *(const float4*)(h + (size_t)s * D_HID + c);
    float* ap = agg + (size_t)d * D_HID + c;
    atomicAdd(ap + 0, hv.x * norm);
    atomicAdd(ap + 1, hv.y * norm);
    atomicAdd(ap + 2, hv.z * norm);
    atomicAdd(ap + 3, hv.w * norm);
}

// h <- relu(agg + h * dinv^2 + bias)   (in place)
__global__ void gcn_finalize(float* __restrict__ h, const float* __restrict__ agg,
                             const float* __restrict__ dinv,
                             const float* __restrict__ bias, int n)
{
    long long i = (long long)blockIdx.x * blockDim.x + threadIdx.x;
    if (i >= (long long)n * D_HID) return;
    int node = (int)(i >> 7);
    int c    = (int)(i & (D_HID - 1));
    float di = dinv[node];
    float v = agg[i] + h[i] * di * di + bias[c];
    h[i] = fmaxf(v, 0.0f);
}

__global__ void row_norm(const float* __restrict__ embs, float* __restrict__ nrm, int n)
{
    int i = blockIdx.x * blockDim.x + threadIdx.x;
    if (i >= n) return;
    const float* r = embs + (size_t)i * BASIS;
    float s = 0.0f;
    #pragma unroll
    for (int k = 0; k < BASIS; ++k) s += r[k] * r[k];
    nrm[i] = sqrtf(s) + 1e-8f;
}

__global__ void batch_count(const int* __restrict__ batch, float* __restrict__ cnt, int n)
{
    int i = blockIdx.x * blockDim.x + threadIdx.x;
    if (i < n) atomicAdd(&cnt[batch[i]], 1.0f);
}

// cosine[n][j] = dot(embs[n], basis[:,j]) / nrm[n];  pooled[g][j] += cosine
__global__ void cosine_pool(const float* __restrict__ embs,
                            const float* __restrict__ basis,
                            const float* __restrict__ nrm,
                            const int* __restrict__ batch,
                            float* __restrict__ cosine,
                            float* __restrict__ pooled, int n)
{
    long long tid = (long long)blockIdx.x * blockDim.x + threadIdx.x;
    int node = (int)(tid >> 6);
    int j    = (int)(tid & 63);
    if (node >= n || j >= KC) return;
    const float* r = embs + (size_t)node * BASIS;
    float acc = 0.0f;
    #pragma unroll
    for (int k = 0; k < BASIS; ++k) acc = fmaf(r[k], basis[k * KC + j], acc);
    float v = acc / nrm[node];
    cosine[(size_t)node * KC + j] = v;
    atomicAdd(&pooled[batch[node] * KC + j], v);
}

__global__ void classify(const float* __restrict__ pooled, const float* __restrict__ cnt,
                         const float* __restrict__ cw, float* __restrict__ out)
{
    int tid = blockIdx.x * blockDim.x + threadIdx.x;
    if (tid >= NGRAPH * NCLS) return;
    int g = tid / NCLS, c = tid % NCLS;
    float inv = 1.0f / fmaxf(cnt[g], 1.0f);
    float acc = 0.0f;
    #pragma unroll
    for (int j = 0; j < KC; ++j) acc = fmaf(pooled[g * KC + j] * inv, cw[j * NCLS + c], acc);
    out[g * NCLS + c] = acc;
}

// ---------------------------------------------------------------------------
// Launcher
// ---------------------------------------------------------------------------
extern "C" void kernel_launch(void* const* d_in, const int* in_sizes, int n_in,
                              void* d_out, int out_size, void* d_ws, size_t ws_size,
                              hipStream_t stream)
{
    const float* x     = (const float*)d_in[0];
    const int*   src   = (const int*)  d_in[1];
    const int*   dst   = (const int*)  d_in[2];
    const int*   batch = (const int*)  d_in[3];
    const float* W1 = (const float*)d_in[4];  const float* b1 = (const float*)d_in[5];
    const float* W2 = (const float*)d_in[6];  const float* b2 = (const float*)d_in[7];
    const float* W3 = (const float*)d_in[8];  const float* b3 = (const float*)d_in[9];
    const float* Wm = (const float*)d_in[10]; const float* bm = (const float*)d_in[11];
    const float* basis = (const float*)d_in[12];
    const float* cw    = (const float*)d_in[13];

    const int N = in_sizes[3];   // nodes (== batch length)
    const int E = in_sizes[1];   // edges

    // output layout: out[G*C] | node_embs[N*64] | cosine[N*50]
    float* out_cls   = (float*)d_out;
    float* node_embs = out_cls + NGRAPH * NCLS;
    float* cosine    = node_embs + (size_t)N * BASIS;

    // workspace layout (floats)
    float* HA     = (float*)d_ws;                       // N*128
    float* HB     = HA  + (size_t)N * D_HID;            // N*128
    float* AGG    = HB  + (size_t)N * D_HID;            // N*128
    float* deg    = AGG + (size_t)N * D_HID;            // N
    float* dinv   = deg + N;                            // N
    float* nrm    = dinv + N;                           // N
    float* pooled = nrm + N;                            // G*KC
    float* cnt    = pooled + NGRAPH * KC;               // G

    const long long nh = (long long)N * D_HID;
    auto blks = [](long long n, int b) { return (unsigned)((n + b - 1) / b); };

    // degrees
    zero_f32<<<blks(N, 256), 256, 0, stream>>>(deg, N);
    deg_count<<<blks(E, 256), 256, 0, stream>>>(dst, deg, E);
    deg_rsqrt<<<blks(N, 256), 256, 0, stream>>>(deg, dinv, N);

    const unsigned gemm_grid = blks(N, 128);   // 128 rows per block (8 waves x 16)

    // --- layer 1: x -> HA ---
    gemm_wmma_f16<128><<<gemm_grid, 256, 0, stream>>>(x,  W1, nullptr, HA, N, 0);
    zero_f32<<<blks(nh, 256), 256, 0, stream>>>(AGG, nh);
    edge_msg<<<blks((long long)E * 32, 256), 256, 0, stream>>>(src, dst, dinv, HA, AGG, E);
    gcn_finalize<<<blks(nh, 256), 256, 0, stream>>>(HA, AGG, dinv, b1, N);

    // --- layer 2: HA -> HB ---
    gemm_wmma_f16<128><<<gemm_grid, 256, 0, stream>>>(HA, W2, nullptr, HB, N, 0);
    zero_f32<<<blks(nh, 256), 256, 0, stream>>>(AGG, nh);
    edge_msg<<<blks((long long)E * 32, 256), 256, 0, stream>>>(src, dst, dinv, HB, AGG, E);
    gcn_finalize<<<blks(nh, 256), 256, 0, stream>>>(HB, AGG, dinv, b2, N);

    // --- layer 3: HB -> HA ---
    gemm_wmma_f16<128><<<gemm_grid, 256, 0, stream>>>(HB, W3, nullptr, HA, N, 0);
    zero_f32<<<blks(nh, 256), 256, 0, stream>>>(AGG, nh);
    edge_msg<<<blks((long long)E * 32, 256), 256, 0, stream>>>(src, dst, dinv, HA, AGG, E);
    gcn_finalize<<<blks(nh, 256), 256, 0, stream>>>(HA, AGG, dinv, b3, N);

    // --- projection: node_embs = HA @ Wm + bm ---
    gemm_wmma_f16<64><<<gemm_grid, 256, 0, stream>>>(HA, Wm, bm, node_embs, N, 0);

    // --- cosine similarities + pooling ---
    row_norm<<<blks(N, 256), 256, 0, stream>>>(node_embs, nrm, N);
    zero_f32<<<blks(NGRAPH * KC + NGRAPH, 256), 256, 0, stream>>>(pooled, NGRAPH * KC + NGRAPH);
    batch_count<<<blks(N, 256), 256, 0, stream>>>(batch, cnt, N);
    cosine_pool<<<blks((long long)N * 64, 256), 256, 0, stream>>>(node_embs, basis, nrm,
                                                                  batch, cosine, pooled, N);
    classify<<<blks(NGRAPH * NCLS, 256), 256, 0, stream>>>(pooled, cnt, cw, out_cls);
}
